// DPConv_63058709839956
// MI455X (gfx1250) — matrix operand order repaired and checked by reading, not compiled, not run on hardware
//
#include <hip/hip_runtime.h>
#include <hip/hip_bf16.h>

// ---------------------------------------------------------------------------
// Types for CDNA5 WMMA (gfx1250, wave32)
// ---------------------------------------------------------------------------
typedef __attribute__((ext_vector_type(16))) __bf16 v16bf;
typedef __attribute__((ext_vector_type(8)))  float  v8f;

union FragU {
    v16bf v;
    uint4 u[2];
};

__device__ __forceinline__ unsigned short f2bf(float f) {
    unsigned int u = __float_as_uint(f);
    unsigned int r = u + 0x7FFFu + ((u >> 16) & 1u);   // round-to-nearest-even
    return (unsigned short)(r >> 16);
}

// A-matrix fragment (16x32 bf16, M x K), row-major source W[row][k], ld elements.
// ISA layout: lane L: m = L&15, h = L>>4; VGPR v<4: K = 8h+2v,+1 ; v>=4: K = 16+8h+2(v-4),+1
__device__ __forceinline__ v16bf load_frag_A(const unsigned short* __restrict__ W,
                                             int ld, int row0, int k0, int lane) {
    int m = lane & 15, hh = lane >> 4;
    const unsigned short* p = W + (size_t)(row0 + m) * ld + k0 + 8 * hh;
    FragU f;
    f.u[0] = *(const uint4*)(p);        // K = 8h + 0..7
    f.u[1] = *(const uint4*)(p + 16);   // K = 16 + 8h + 0..7
    return f.v;
}

// B-matrix fragment (32x16 bf16, K x N) from K-major-contiguous source Y[n][k].
// ISA layout: lane L: n = L&15, kh = L>>4; VGPR v<4: K = 8kh+2v,+1 ; v>=4: K = 16+8kh+2(v-4),+1
__device__ __forceinline__ v16bf load_frag_B(const unsigned short* __restrict__ Y,
                                             int ld, int n0, int k0, int lane) {
    int n = lane & 15, kh = lane >> 4;
    const unsigned short* p = Y + (size_t)(n0 + n) * ld + k0 + 8 * kh;
    FragU f;
    f.u[0] = *(const uint4*)(p);        // K = 8kh + 0..7
    f.u[1] = *(const uint4*)(p + 16);   // K = 16 + 8kh + 0..7
    return f.v;
}

__device__ __forceinline__ v8f wmma_bf16(v16bf a, v16bf b, v8f c) {
    return __builtin_amdgcn_wmma_f32_16x16x32_bf16(false, a, false, b, (short)0, c,
                                                   false, false);
}

// ---------------------------------------------------------------------------
// Problem constants
// ---------------------------------------------------------------------------
#define NIMG   8
#define CCH    256
#define HW     64
#define NHEAD  4
#define KEYD   32
#define HEADD  64
#define SCALE  0.17677669529663687f   // 32^-0.5

// ---------------------------------------------------------------------------
// Kernel 0: convert weights to bf16 (then L2-resident for all patches)
// ---------------------------------------------------------------------------
__global__ void convert_w_kernel(const float* __restrict__ qkvw,
                                 const float* __restrict__ projw,
                                 unsigned short* __restrict__ wq,
                                 unsigned short* __restrict__ wp) {
    int i = blockIdx.x * 256 + threadIdx.x;
    if (i < 512 * 256) wq[i] = f2bf(qkvw[i]);
    if (i < 256 * 256) wp[i] = f2bf(projw[i]);
}

// Kernel 0b: out = x / 3  (fold-scatter accumulates on top, pre-scaled)
__global__ void init_out_kernel(const float* __restrict__ x,
                                float* __restrict__ out, int n) {
    int i = blockIdx.x * blockDim.x + threadIdx.x;
    if (i < n) out[i] = x[i] * (1.0f / 3.0f);
}

// ---------------------------------------------------------------------------
// Kernel 1: pooling (2-tap avg, stride 4 patches, edge pad) + depthwise 3x3 +
// residual, all per-channel in registers. One block per patch, thread = channel.
// Output y bf16, token-major: ybuf[(patch*T + t)*256 + c]
// ---------------------------------------------------------------------------
template <int K, int L>
__global__ __launch_bounds__(256) void pool_dw_kernel(
    const float* __restrict__ x, const float* __restrict__ dw,
    const float* __restrict__ db, unsigned short* __restrict__ ybuf) {
    constexpr int Kp = K + 2;
    constexpr int T = K * K;
    const int c = threadIdx.x;
    const int b = blockIdx.x;
    const int n  = b / (L * L);
    const int lh = (b / L) % L;
    const int lw = b % L;
    const float* xc = x + ((size_t)(n * CCH + c)) * HW * HW;

    float w9[9];
#pragma unroll
    for (int i = 0; i < 9; ++i) w9[i] = dw[c * 9 + i];
    const float bias = db[c];

    float pooled[K][K];
#pragma unroll
    for (int p = 0; p < K; ++p)
#pragma unroll
        for (int q = 0; q < K; ++q) pooled[p][q] = 0.f;

#pragma unroll
    for (int i = 0; i < Kp; ++i) {
        int hp = lh * 4 + i;                 // padded coord 0..65
        int hh = min(max(hp - 1, 0), HW - 1);
        float vals[Kp];
#pragma unroll
        for (int j = 0; j < Kp; ++j) {
            int wp_ = lw * 4 + j;
            int ww = min(max(wp_ - 1, 0), HW - 1);
            vals[j] = xc[hh * HW + ww];
        }
        float rowp[K];
#pragma unroll
        for (int q = 0; q < K; ++q) {
            int j0 = q * Kp / K;             // pool rows always span exactly 2 cols
            rowp[q] = 0.5f * (vals[j0] + vals[j0 + 1]);
        }
#pragma unroll
        for (int p = 0; p < K; ++p) {
            int i0 = p * Kp / K;
            if (i == i0 || i == i0 + 1) {
#pragma unroll
                for (int q = 0; q < K; ++q) pooled[p][q] += 0.5f * rowp[q];
            }
        }
    }

    unsigned short* yb = ybuf + (size_t)b * T * CCH + c;
#pragma unroll
    for (int p = 0; p < K; ++p)
#pragma unroll
        for (int q = 0; q < K; ++q) {
            float acc = pooled[p][q] + bias;   // residual + conv bias
#pragma unroll
            for (int ky = 0; ky < 3; ++ky) {
                int pp = p + ky - 1;
                if (pp < 0 || pp >= K) continue;
#pragma unroll
                for (int kx = 0; kx < 3; ++kx) {
                    int qq = q + kx - 1;
                    if (qq < 0 || qq >= K) continue;
                    acc += w9[ky * 3 + kx] * pooled[pp][qq];
                }
            }
            yb[(size_t)(p * K + q) * CCH] = f2bf(acc);
        }
}

// ---------------------------------------------------------------------------
// Kernel 2: qkv GEMM (WMMA bf16) + softmax attention + PE depthwise conv on v.
// One block per (patch, head), 128 threads = 4 waves.
// ---------------------------------------------------------------------------
template <int K>
__global__ __launch_bounds__(128) void attn_kernel(
    const unsigned short* __restrict__ ybuf,
    const unsigned short* __restrict__ wq,      // 512x256 bf16 row-major
    const float* __restrict__ qkv_b,
    const float* __restrict__ pe_w, const float* __restrict__ pe_b,
    unsigned short* __restrict__ obuf) {
    constexpr int T  = K * K;        // tokens: 16 or 64
    constexpr int NT = T / 16;       // N tiles: 1 or 4
    constexpr int JOBS = 8 * NT;     // 8 M-tiles (128 rows per head)
    constexpr int JPW  = JOBS / 4;

    __shared__ float qs[32 * T];
    __shared__ float ks[32 * T];
    __shared__ float vs[64 * T];
    __shared__ float Sm[T * T];

    const int tid = threadIdx.x, lane = tid & 31, wv = tid >> 5;
    const int b = blockIdx.x, h = blockIdx.y;
    const unsigned short* Y = ybuf + (size_t)b * T * CCH;

    // ---- qkv = W[h*128 .. h*128+127] * y  (M=128, N=T, K=256) ----
    for (int j = 0; j < JPW; ++j) {
        int job = wv * JPW + j;
        int mt = job / NT, nt = job % NT;
        int row0 = h * 128 + mt * 16;
        v8f acc = {0.f, 0.f, 0.f, 0.f, 0.f, 0.f, 0.f, 0.f};
#pragma unroll
        for (int ko = 0; ko < 8; ++ko) {
            if (ko < 7)
                __builtin_prefetch(wq + (size_t)row0 * CCH + (ko + 1) * 32, 0, 0);
            v16bf a  = load_frag_A(wq, CCH, row0, ko * 32, lane);
            v16bf bb = load_frag_B(Y, CCH, nt * 16, ko * 32, lane);
            acc = wmma_bf16(a, bb, acc);
        }
        int nn = lane & 15, hh = lane >> 4;
        int t = nt * 16 + nn;
#pragma unroll
        for (int r = 0; r < 8; ++r) {
            int rl = mt * 16 + r + 8 * hh;           // 0..127 within head
            float val = acc[r] + qkv_b[h * 128 + rl];
            if (rl < 32)       qs[rl * T + t] = val;
            else if (rl < 64)  ks[(rl - 32) * T + t] = val;
            else               vs[(rl - 64) * T + t] = val;
        }
    }
    __syncthreads();

    // ---- S[n][m] = scale * q[:,n].k[:,m] ----
    for (int idx = tid; idx < T * T; idx += 128) {
        int n = idx / T, m = idx % T;
        float s = 0.f;
#pragma unroll 8
        for (int d = 0; d < KEYD; ++d) s += qs[d * T + n] * ks[d * T + m];
        Sm[idx] = s * SCALE;
    }
    __syncthreads();

    // ---- row softmax ----
    if (tid < T) {
        float mx = -3.4e38f;
        for (int m = 0; m < T; ++m) mx = fmaxf(mx, Sm[tid * T + m]);
        float sum = 0.f;
        for (int m = 0; m < T; ++m) {
            float e = __expf(Sm[tid * T + m] - mx);
            Sm[tid * T + m] = e;
            sum += e;
        }
        float inv = 1.f / sum;
        for (int m = 0; m < T; ++m) Sm[tid * T + m] *= inv;
    }
    __syncthreads();

    // ---- O[d][t] = v·attn + PE-dwconv(v) + pe_b ; store bf16 [t][c] ----
    for (int idx = tid; idx < HEADD * T; idx += 128) {
        int d = idx / T, t = idx % T;
        float o = 0.f;
#pragma unroll 8
        for (int m = 0; m < T; ++m) o += vs[d * T + m] * Sm[t * T + m];
        int c = h * HEADD + d;
        int p = t / K, q = t % K;
        float cv = pe_b[c];
#pragma unroll
        for (int ky = 0; ky < 3; ++ky) {
            int pp = p + ky - 1;
            if (pp < 0 || pp >= K) continue;
#pragma unroll
            for (int kx = 0; kx < 3; ++kx) {
                int qq = q + kx - 1;
                if (qq < 0 || qq >= K) continue;
                cv += pe_w[c * 9 + ky * 3 + kx] * vs[d * T + pp * K + qq];
            }
        }
        obuf[(size_t)(b * T + t) * CCH + c] = f2bf(o + cv);
    }
}

// ---------------------------------------------------------------------------
// Kernel 3: proj GEMM (WMMA bf16) + fold-scatter (raw-reshape semantics) with
// analytic overlap counts, accumulating val/(3*cntH*cntW) into d_out.
// One block per patch, 256 threads = 8 waves.
// ---------------------------------------------------------------------------
template <int K>
__device__ __forceinline__ float invcnt(int h) {
    if (K == 8) return (h >= 4 && h < 60) ? 0.5f : 1.0f;
    return 1.0f;   // K = 4: exactly one contribution everywhere
}

template <int K, int L>
__global__ __launch_bounds__(256) void proj_fold_kernel(
    const unsigned short* __restrict__ obuf,
    const unsigned short* __restrict__ wp,      // 256x256 bf16 row-major
    const float* __restrict__ proj_b,
    float* __restrict__ out) {
    constexpr int T  = K * K;
    constexpr int NT = T / 16;
    constexpr int JOBS = 16 * NT;    // 16 M-tiles of 256 output channels
    constexpr int JPW  = JOBS / 8;

    const int tid = threadIdx.x, lane = tid & 31, wv = tid >> 5;
    const int b = blockIdx.x;
    const int n_img = b / (L * L);
    const int pi = b % (L * L);      // lh*L + lw
    const unsigned short* O = obuf + (size_t)b * T * CCH;

    for (int j = 0; j < JPW; ++j) {
        int job = wv * JPW + j;
        int mt = job / NT, nt = job % NT;
        v8f acc = {0.f, 0.f, 0.f, 0.f, 0.f, 0.f, 0.f, 0.f};
#pragma unroll
        for (int ko = 0; ko < 8; ++ko) {
            if (ko < 7)
                __builtin_prefetch(wp + (size_t)mt * 16 * CCH + (ko + 1) * 32, 0, 0);
            v16bf a  = load_frag_A(wp, CCH, mt * 16, ko * 32, lane);
            v16bf bb = load_frag_B(O, CCH, nt * 16, ko * 32, lane);
            acc = wmma_bf16(a, bb, acc);
        }
        int nn = lane & 15, hh = lane >> 4;
        int t = nt * 16 + nn;
        int p = t / K, q = t % K;
#pragma unroll
        for (int r = 0; r < 8; ++r) {
            int orow = mt * 16 + r + 8 * hh;         // output channel 0..255
            float val = acc[r] + proj_b[orow];
            // flat offset inside image: reference does a raw reshape
            // (Lh,Lw,C,K,K) -> (C,K,K,Lh,Lw), so decompose the flat index.
            int f = ((pi * CCH + orow) * K + p) * K + q;
            int l2 = f % L;  f /= L;
            int l1 = f % L;  f /= L;
            int k2 = f % K;  f /= K;
            int k1 = f % K;
            int c  = f / K;
            int hc = l1 * 4 + k1;
            int wc = l2 * 4 + k2;
            float scale = (1.0f / 3.0f) * invcnt<K>(hc) * invcnt<K>(wc);
            atomicAdd(&out[((size_t)(n_img * CCH + c) * HW + hc) * HW + wc],
                      val * scale);
        }
    }
}

// ---------------------------------------------------------------------------
// Launch
// ---------------------------------------------------------------------------
extern "C" void kernel_launch(void* const* d_in, const int* in_sizes, int n_in,
                              void* d_out, int out_size, void* d_ws, size_t ws_size,
                              hipStream_t stream) {
    const float* x         = (const float*)d_in[0];
    const float* qkv_w     = (const float*)d_in[1];
    const float* qkv_b     = (const float*)d_in[2];
    const float* proj_w    = (const float*)d_in[3];
    const float* proj_b    = (const float*)d_in[4];
    const float* attn_pe_w = (const float*)d_in[5];
    const float* attn_pe_b = (const float*)d_in[6];
    const float* dp_pe_w   = (const float*)d_in[7];
    const float* dp_pe_b   = (const float*)d_in[8];
    float* out = (float*)d_out;

    // workspace layout (bytes):
    //   [0, 256K)       qkv_w bf16 (512x256)
    //   [256K, 384K)    proj_w bf16 (256x256)
    //   [384K, +59MB)   y buffer bf16 (reused K=4 then K=8)
    //   [.., +59MB)     o buffer bf16 (reused K=4 then K=8)
    unsigned short* wq   = (unsigned short*)d_ws;
    unsigned short* wpj  = wq + 512 * 256;
    unsigned short* ybuf = (unsigned short*)((char*)d_ws + 393216);
    unsigned short* obuf = (unsigned short*)((char*)d_ws + 393216 + 59000000);

    convert_w_kernel<<<512, 256, 0, stream>>>(qkv_w, proj_w, wq, wpj);

    const int nout = NIMG * CCH * HW * HW;   // 8.39M
    init_out_kernel<<<(nout + 255) / 256, 256, 0, stream>>>(x, out, nout);

    // ---- K = 4 : L = 16, 2048 patches, 16 tokens ----
    pool_dw_kernel<4, 16><<<2048, 256, 0, stream>>>(x, dp_pe_w, dp_pe_b, ybuf);
    attn_kernel<4><<<dim3(2048, 4), 128, 0, stream>>>(ybuf, wq, qkv_b,
                                                      attn_pe_w, attn_pe_b, obuf);
    proj_fold_kernel<4, 16><<<2048, 256, 0, stream>>>(obuf, wpj, proj_b, out);

    // ---- K = 8 : L = 15, 1800 patches, 64 tokens ----
    pool_dw_kernel<8, 15><<<1800, 256, 0, stream>>>(x, dp_pe_w, dp_pe_b, ybuf);
    attn_kernel<8><<<dim3(1800, 4), 128, 0, stream>>>(ybuf, wq, qkv_b,
                                                      attn_pe_w, attn_pe_b, obuf);
    proj_fold_kernel<8, 15><<<1800, 256, 0, stream>>>(obuf, wpj, proj_b, out);
}
